// GCNGraphEmbedding_70875550319263
// MI455X (gfx1250) — matrix operand order repaired and checked by reading, not compiled, not run on hardware
//
#include <hip/hip_runtime.h>
#include <hip/hip_bf16.h>

typedef __attribute__((ext_vector_type(16))) _Float16 v16h;
typedef __attribute__((ext_vector_type(8)))  _Float16 v8h;
typedef __attribute__((ext_vector_type(8)))  float    v8f;

#define HFEAT 128
#define GRAPHS 64
#define GEMM_WAVES 4                  // 128-thread blocks -> 4 wave32
#define GEMM_ROWS (GEMM_WAVES * 16)   // 64 rows per block
#define LDS_STRIDE 136                // 272B row stride: bank step 4, 16B aligned

// gemm flags
#define GF_BIAS_C  1   // C += bias (final head)
#define GF_RELU_IN 2   // relu on A during f16 staging (layer 2)
#define GF_WRITE_H 4   // also write H = acc*dinv(row)^2 + bias (self-loop init)

// ---------------------------------------------------------------- utilities
__global__ void fill_kernel(float* __restrict__ p, float v, int n) {
    int i = blockIdx.x * blockDim.x + threadIdx.x;
    if (i < n) p[i] = v;
}

__global__ void degree_kernel(const int* __restrict__ dst, float* __restrict__ deg, int E) {
    int e = blockIdx.x * blockDim.x + threadIdx.x;
    if (e < E) unsafeAtomicAdd(&deg[dst[e]], 1.0f);
}

__global__ void rsqrt_kernel(float* __restrict__ deg, int N) {
    int i = blockIdx.x * blockDim.x + threadIdx.x;
    if (i < N) {
        float d = deg[i];
        deg[i] = (d > 0.0f) ? rsqrtf(d) : 0.0f;   // becomes dinv in place
    }
}

__global__ void norm_kernel(const int* __restrict__ src, const int* __restrict__ dst,
                            const float* __restrict__ dinv, float* __restrict__ norm, int E) {
    int e = blockIdx.x * blockDim.x + threadIdx.x;
    if (e < E) norm[e] = dinv[src[e]] * dinv[dst[e]];
}

// one thread per (edge, 4-feature chunk): b128 gather + 4 f32 HW atomics (L2-resident)
__global__ void scatter_kernel(const float* __restrict__ xw, const int* __restrict__ src,
                               const int* __restrict__ dst, const float* __restrict__ norm,
                               float* __restrict__ h, int E) {
    long long i = (long long)blockIdx.x * blockDim.x + threadIdx.x;
    long long total = (long long)E * (HFEAT / 4);
    if (i >= total) return;
    int e = (int)(i >> 5);
    int c = (int)(i & 31);
    float4 v = ((const float4*)xw)[(size_t)src[e] * 32 + c];
    float w = norm[e];
    float* hp = &h[(size_t)dst[e] * HFEAT + c * 4];
    unsafeAtomicAdd(hp + 0, v.x * w);
    unsafeAtomicAdd(hp + 1, v.y * w);
    unsafeAtomicAdd(hp + 2, v.z * w);
    unsafeAtomicAdd(hp + 3, v.w * w);
}

__global__ void count_kernel(const int* __restrict__ batch, float* __restrict__ cnt, int N) {
    int n = blockIdx.x * blockDim.x + threadIdx.x;
    if (n < N) unsafeAtomicAdd(&cnt[batch[n]], 1.0f);
}

// batch is sorted: thread f keeps a register sum, flushes one atomic per graph
// boundary. ReLU of layer-2 output fused into the accumulation.
#define POOL_NODES 128
__global__ void pool_sum_kernel(const float* __restrict__ h, const int* __restrict__ batch,
                                float* __restrict__ sums, int N) {
    int f = threadIdx.x;                 // 128 threads
    int n0 = blockIdx.x * POOL_NODES;
    if (n0 >= N) return;
    int n1 = min(n0 + POOL_NODES, N);
    float acc = 0.0f;
    int gcur = batch[n0];
    for (int n = n0; n < n1; ++n) {
        int g = batch[n];
        if (g != gcur) {
            unsafeAtomicAdd(&sums[gcur * HFEAT + f], acc);
            acc = 0.0f; gcur = g;
        }
        acc += fmaxf(h[(size_t)n * HFEAT + f], 0.0f);   // fused ReLU
    }
    unsafeAtomicAdd(&sums[gcur * HFEAT + f], acc);
}

__global__ void divide_kernel(const float* __restrict__ sums, const float* __restrict__ cnt,
                              float* __restrict__ pooled, int total) {
    int i = blockIdx.x * blockDim.x + threadIdx.x;
    if (i < total) {
        int g = i >> 7;
        pooled[i] = sums[i] / fmaxf(cnt[g], 1.0f);
    }
}

// ---------------------------------------------------------------- WMMA GEMM
// C[M,128] = A[M,128] @ W[128,128], f32 in/out, f16 WMMA, f32 accum.
// flags: GF_BIAS_C  -> C += bias
//        GF_RELU_IN -> relu(A) during staging
//        GF_WRITE_H -> H = acc*dinv(row)^2 + bias  (GCN self-loop + bias init)
// H may alias A: each block stages only its own rows before the barrier and
// writes only those rows after it.
__global__ void __launch_bounds__(128)
gemm128_wmma_kernel(const float* __restrict__ A, const float* __restrict__ W,
                    float* __restrict__ C, int M,
                    const float* __restrict__ bias, const float* __restrict__ dinv,
                    float* __restrict__ H, int flags) {
    __shared__ _Float16 WT[HFEAT][LDS_STRIDE];                 // W transposed: WT[n][k]
    __shared__ _Float16 XS[GEMM_WAVES][16][LDS_STRIDE];        // per-wave A rows (f16)

    const int tid  = threadIdx.x;
    const int wave = tid >> 5;
    const int lane = tid & 31;
    const int row_base = blockIdx.x * GEMM_ROWS + wave * 16;

    // Stage W -> WT (transposed, f16). Reads coalesced over n.
    for (int i = 0; i < 128; ++i) {
        int idx = tid + i * 128;            // 16384 elements
        int k = idx >> 7, n = idx & 127;
        WT[n][k] = (_Float16)W[idx];
    }
    // Stage this wave's 16 A rows (float4-coalesced), optional fused ReLU.
    const bool rin = (flags & GF_RELU_IN) != 0;
    for (int j = 0; j < 16; ++j) {
        int r = row_base + j;
        int rc = (r < M) ? r : (M - 1);
        float4 v = ((const float4*)A)[(size_t)rc * 32 + lane];
        if (rin) {
            v.x = fmaxf(v.x, 0.f); v.y = fmaxf(v.y, 0.f);
            v.z = fmaxf(v.z, 0.f); v.w = fmaxf(v.w, 0.f);
        }
        _Float16* dsti = &XS[wave][j][lane * 4];
        dsti[0] = (_Float16)v.x; dsti[1] = (_Float16)v.y;
        dsti[2] = (_Float16)v.z; dsti[3] = (_Float16)v.w;
    }
    __syncthreads();

    const int m     = lane & 15;
    const int half  = lane >> 4;       // 0 or 1
    const int khalf = half << 4;       // A: +16 k for upper half-wave
    const int koff2 = half << 4;       // B: +16 k for upper half-wave
    const int ncol  = lane & 15;

    v8f acc[8];
#pragma unroll
    for (int t = 0; t < 8; ++t) acc[t] = (v8f){0.f,0.f,0.f,0.f,0.f,0.f,0.f,0.f};

#pragma unroll
    for (int kk = 0; kk < 128; kk += 32) {
        // A fragment 16x32 (contiguous k = 16*half + i, matching B): two contiguous 16B LDS loads
        v8h alo = *(const v8h*)&XS[wave][m][kk + khalf];
        v8h ahi = *(const v8h*)&XS[wave][m][kk + khalf + 8];
        v16h a;
#pragma unroll
        for (int i = 0; i < 8; ++i) { a[i] = alo[i]; a[i + 8] = ahi[i]; }

#pragma unroll
        for (int t = 0; t < 8; ++t) {
            const _Float16* pb = &WT[t * 16 + ncol][kk + koff2];
            v8h blo = *(const v8h*)pb;
            v8h bhi = *(const v8h*)(pb + 8);
            v16h b;
#pragma unroll
            for (int i = 0; i < 8; ++i) { b[i] = blo[i]; b[i + 8] = bhi[i]; }
            acc[t] = __builtin_amdgcn_wmma_f32_16x16x32_f16(
                false, a, false, b, (short)0, acc[t], false, false);
        }
    }

    // D layout: VGPR r -> M = r + 8*half, N = ncol. Per (r,t) store: lanes 0-15
    // and 16-31 each cover a contiguous 64B segment.
    float bv[8];
#pragma unroll
    for (int t = 0; t < 8; ++t) bv[t] = bias[t * 16 + ncol];

    const int  mofs  = half << 3;
    const bool addbc = (flags & GF_BIAS_C) != 0;
    const bool wh    = (flags & GF_WRITE_H) != 0;
#pragma unroll
    for (int r = 0; r < 8; ++r) {
        int grow = row_base + r + mofs;
        if (grow >= M) continue;
        float di2 = 0.0f;
        if (wh) { float di = dinv[grow]; di2 = di * di; }
#pragma unroll
        for (int t = 0; t < 8; ++t) {
            int n = t * 16 + ncol;
            float val = acc[t][r];
            C[(size_t)grow * HFEAT + n] = addbc ? (val + bv[t]) : val;
            if (wh) H[(size_t)grow * HFEAT + n] = val * di2 + bv[t];
        }
    }
}

// ---------------------------------------------------------------- launcher
extern "C" void kernel_launch(void* const* d_in, const int* in_sizes, int n_in,
                              void* d_out, int out_size, void* d_ws, size_t ws_size,
                              hipStream_t stream) {
    const float* x   = (const float*)d_in[0];
    const int*   ei  = (const int*)d_in[1];
    const int*   bat = (const int*)d_in[2];
    const float* W1  = (const float*)d_in[3];
    const float* b1  = (const float*)d_in[4];
    const float* W2  = (const float*)d_in[5];
    const float* b2  = (const float*)d_in[6];
    const float* Wl  = (const float*)d_in[7];
    const float* bl  = (const float*)d_in[8];
    float* out = (float*)d_out;

    const int N = in_sizes[0] / HFEAT;
    const int E = in_sizes[1] / 2;
    const int* src = ei;
    const int* dst = ei + E;

    // workspace layout (f32)
    float* xw   = (float*)d_ws;                    // N*128
    float* h    = xw + (size_t)N * HFEAT;          // N*128
    float* dinv = h + (size_t)N * HFEAT;           // N  (deg first, dinv after rsqrt)
    float* nrm  = dinv + N;                        // E
    float* sums = nrm + E;                         // 64*128
    float* cnt  = sums + GRAPHS * HFEAT;           // 64
    float* pool = cnt + GRAPHS;                    // 64*128

    const int TB = 256;
    const int gemm_grid = (N + GEMM_ROWS - 1) / GEMM_ROWS;
    const long long etot = (long long)E * (HFEAT / 4);
    const int egrid = (int)((etot + TB - 1) / TB);

    // degrees -> dinv -> per-edge norms
    fill_kernel<<<(N + TB - 1) / TB, TB, 0, stream>>>(dinv, 1.0f, N);  // self loop
    degree_kernel<<<(E + TB - 1) / TB, TB, 0, stream>>>(dst, dinv, E);
    rsqrt_kernel<<<(N + TB - 1) / TB, TB, 0, stream>>>(dinv, N);
    norm_kernel<<<(E + TB - 1) / TB, TB, 0, stream>>>(src, dst, dinv, nrm, E);

    // ---- layer 1: xw1 = x@W1 ; h = xw1*dinv^2 + b1 (fused) ; edge scatter
    gemm128_wmma_kernel<<<gemm_grid, 128, 0, stream>>>(x, W1, xw, N, b1, dinv, h, GF_WRITE_H);
    scatter_kernel<<<egrid, TB, 0, stream>>>(xw, src, dst, nrm, h, E);

    // ---- layer 2: relu(h1) fused into staging; h aliases A (safe, see kernel)
    gemm128_wmma_kernel<<<gemm_grid, 128, 0, stream>>>(h, W2, xw, N, b2, dinv, h,
                                                       GF_RELU_IN | GF_WRITE_H);
    scatter_kernel<<<egrid, TB, 0, stream>>>(xw, src, dst, nrm, h, E);

    // ---- mean pool over sorted batch (relu(h2) fused into accumulation)
    fill_kernel<<<(GRAPHS * HFEAT + TB - 1) / TB, TB, 0, stream>>>(sums, 0.0f, GRAPHS * HFEAT);
    fill_kernel<<<1, GRAPHS, 0, stream>>>(cnt, 0.0f, GRAPHS);
    count_kernel<<<(N + TB - 1) / TB, TB, 0, stream>>>(bat, cnt, N);
    pool_sum_kernel<<<(N + POOL_NODES - 1) / POOL_NODES, HFEAT, 0, stream>>>(h, bat, sums, N);
    divide_kernel<<<(GRAPHS * HFEAT + TB - 1) / TB, TB, 0, stream>>>(sums, cnt, pool, GRAPHS * HFEAT);

    // ---- linear head: [64,128] @ [128,128] + bl -> d_out (single block)
    gemm128_wmma_kernel<<<1, 128, 0, stream>>>(pool, Wl, out, GRAPHS, bl, dinv, out, GF_BIAS_C);
}
